// Encoder_18305150616327
// MI455X (gfx1250) — matrix operand order repaired and checked by reading, not compiled, not run on hardware
//
#include <hip/hip_runtime.h>
#include <math.h>

#define NN 50000
#define EE 800000
#define FF 128
#define DD 64
#define GG 512
#define TD 192   // 3*D

typedef __attribute__((ext_vector_type(16))) __bf16 v16bf;
typedef __attribute__((ext_vector_type(8)))  float  v8f;

static __device__ __forceinline__ __bf16 f2bf(float x) { return (__bf16)x; }

// ----------------------------- utility kernels -----------------------------
__global__ void fill_kernel(float* p, float v, int n) {
  int t = blockIdx.x * blockDim.x + threadIdx.x;
  if (t < n) p[t] = v;
}

__global__ void copy4_kernel(const float* __restrict__ s, float* __restrict__ d, int n4) {
  int t = blockIdx.x * blockDim.x + threadIdx.x;
  if (t < n4) ((float4*)d)[t] = ((const float4*)s)[t];
}

// ------------------------ edge gather + scatter-add ------------------------
__global__ void edge_scatter_kernel(const float* __restrict__ h, const int* __restrict__ src,
                                    const int* __restrict__ dst, float* agg, int K) {
  int chunks = K >> 2;
  long long t = (long long)blockIdx.x * blockDim.x + threadIdx.x;
  long long total = (long long)EE * chunks;
  if (t >= total) return;
  int e = (int)(t / chunks);
  int c = (int)(t - (long long)e * chunks) << 2;
  int s = src[e], d = dst[e];
  float4 val = *(const float4*)(h + (size_t)s * K + c);
  float* base = agg + (size_t)d * K + c;
  atomicAdd(base + 0, val.x);
  atomicAdd(base + 1, val.y);
  atomicAdd(base + 2, val.z);
  atomicAdd(base + 3, val.w);
}

// -------------------- WMMA GEMM: out[M,64] = act(A[M,K]@W[K,64] + bias) ----
// one wave32 per 16-row tile, 8 waves per block.
// W is staged to LDS pre-swizzled into B-fragment order:
//   element ((ks*4+nt)*32 + lane)*16 + j  ==  bf16(W[(ks*32 + (lane>>4)*8 + (j&7) + (j>>3)*16)*64 + nt*16 + (lane&15)])
// so each lane reads its whole 16-elem fragment as one contiguous 32B LDS access.
__global__ void __launch_bounds__(256)
gemm64_kernel(const float* __restrict__ A, const float* __restrict__ W,
              const float* __restrict__ bias, float* __restrict__ out,
              int M, int K, int act) {
  __shared__ __align__(32) __bf16 Wsw[FF * DD];   // up to 128x64 bf16 = 16KB
  int tid = threadIdx.x;
  int nW = K * DD;
  for (int e = tid; e < nW; e += blockDim.x) {
    int j  = e & 15;
    int l  = (e >> 4) & 31;
    int nt = (e >> 9) & 3;
    int ks = e >> 11;
    int k  = ks * 32 + ((l >> 4) << 3) + (j & 7) + ((j >> 3) << 4);
    int c  = nt * 16 + (l & 15);
    Wsw[e] = f2bf(W[(size_t)k * DD + c]);
  }
  __syncthreads();

  int wave = tid >> 5;
  int lane = tid & 31;
  int tile = blockIdx.x * 8 + wave;
  int m0 = tile * 16;
  if (m0 >= M) return;

  int half = lane >> 4;      // 0 or 1
  int col  = lane & 15;      // also row-within-tile for A

  v8f acc[4] = {v8f{}, v8f{}, v8f{}, v8f{}};
  const float* arow = A + (size_t)(m0 + col) * K;
  const v16bf* Bfrags = (const v16bf*)Wsw;

  for (int ks = 0; ks < (K >> 5); ++ks) {
    // A fragment (16x32 bf16): lane needs K = [kb..kb+7] and [kb+16..kb+23]
    int kb = ks * 32 + half * 8;
    float4 p0 = *(const float4*)(arow + kb);
    float4 p1 = *(const float4*)(arow + kb + 4);
    float4 p2 = *(const float4*)(arow + kb + 16);
    float4 p3 = *(const float4*)(arow + kb + 20);
    v16bf a;
    a[0]  = f2bf(p0.x); a[1]  = f2bf(p0.y); a[2]  = f2bf(p0.z); a[3]  = f2bf(p0.w);
    a[4]  = f2bf(p1.x); a[5]  = f2bf(p1.y); a[6]  = f2bf(p1.z); a[7]  = f2bf(p1.w);
    a[8]  = f2bf(p2.x); a[9]  = f2bf(p2.y); a[10] = f2bf(p2.z); a[11] = f2bf(p2.w);
    a[12] = f2bf(p3.x); a[13] = f2bf(p3.y); a[14] = f2bf(p3.z); a[15] = f2bf(p3.w);
#pragma unroll
    for (int nt = 0; nt < 4; ++nt) {
      v16bf b = Bfrags[(ks * 4 + nt) * 32 + lane];   // one contiguous 32B LDS read
      acc[nt] = __builtin_amdgcn_wmma_f32_16x16x32_bf16(
          false, a, false, b, (short)0, acc[nt], false, false);
    }
  }
  // D fragment: VGPR r -> row m0 + half*8 + r, col = nt*16 + (lane&15)
#pragma unroll
  for (int nt = 0; nt < 4; ++nt) {
    int ccol = nt * 16 + col;
    float bv = bias ? bias[ccol] : 0.0f;
#pragma unroll
    for (int r = 0; r < 8; ++r) {
      float v = acc[nt][r] + bv;
      if (act) v = fmaxf(v, 0.0f);
      out[(size_t)(m0 + half * 8 + r) * DD + ccol] = v;
    }
  }
}

// ----------------------------- BatchNorm -----------------------------------
// stats[0..63]=sum(relu(z)), stats[64..127]=sumsq(relu(z)) over M rows
__global__ void __launch_bounds__(256)
bn_reduce_kernel(const float* __restrict__ z, float* stats, int M) {
  int tid = threadIdx.x;
  int col = tid & 63;
  int sub = tid >> 6;             // 0..3
  float s = 0.f, ss = 0.f;
  for (int m = blockIdx.x * 4 + sub; m < M; m += gridDim.x * 4) {
    float v = fmaxf(z[(size_t)m * DD + col], 0.f);
    s += v; ss += v * v;
  }
  __shared__ float ls[256], lss[256];
  ls[tid] = s; lss[tid] = ss;
  __syncthreads();
  if (sub == 0) {
    s  = ls[col]  + ls[64+col]  + ls[128+col]  + ls[192+col];
    ss = lss[col] + lss[64+col] + lss[128+col] + lss[192+col];
    atomicAdd(&stats[col], s);
    atomicAdd(&stats[64 + col], ss);
  }
}

__global__ void bn_apply_kernel(const float* __restrict__ z, const float* __restrict__ stats,
                                const float* __restrict__ gamma, const float* __restrict__ beta,
                                float* __restrict__ out, int M) {
  int t = blockIdx.x * blockDim.x + threadIdx.x;
  if (t >= M * DD) return;
  int col = t & 63;
  float invM = 1.0f / (float)M;
  float mean = stats[col] * invM;
  float var  = stats[64 + col] * invM - mean * mean;
  float v = fmaxf(z[t], 0.f);
  out[t] = (v - mean) * rsqrtf(var + 1e-5f) * gamma[col] + beta[col];
}

// ----------------------------- summary maker -------------------------------
__global__ void count_nodes_kernel(const int* __restrict__ batch, float* cnt) {
  int n = blockIdx.x * blockDim.x + threadIdx.x;
  if (n < NN) atomicAdd(&cnt[batch[n]], 1.0f);
}

__global__ void slot_sum_kernel(const float* __restrict__ h, const int* __restrict__ batch,
                                float* slots) {
  int t = blockIdx.x * blockDim.x + threadIdx.x;   // NN*16 threads, float4 chunks
  if (t >= NN * 16) return;
  int n = t >> 4;
  int c = (t & 15) << 2;
  int b = batch[n];
  float4 v = *(const float4*)(h + (size_t)n * DD + c);
  float* base = slots + (size_t)b * DD + c;
  atomicAdd(base + 0, v.x); atomicAdd(base + 1, v.y);
  atomicAdd(base + 2, v.z); atomicAdd(base + 3, v.w);
}

__global__ void slot_div_kernel(float* slots, const float* __restrict__ cnt) {
  int t = blockIdx.x * blockDim.x + threadIdx.x;
  if (t >= GG * DD) return;
  slots[t] /= fmaxf(cnt[t >> 6], 1.0f);
}

__global__ void attn_logits_kernel(const float* __restrict__ k, const float* __restrict__ q,
                                   const int* __restrict__ batch, float* __restrict__ logits) {
  int n = blockIdx.x * blockDim.x + threadIdx.x;
  if (n >= NN) return;
  int b = batch[n];
  const float* kr = k + (size_t)n * DD;
  const float* qr = q + (size_t)b * DD;
  float acc = 0.f;
#pragma unroll 8
  for (int c = 0; c < DD; ++c) acc += kr[c] * qr[c];
  logits[n] = acc * 0.125f;   // 1/sqrt(64)
}

static __device__ void atomicMaxF(float* addr, float val) {
  int* ia = (int*)addr;
  int old = __float_as_int(*addr);
  while (__int_as_float(old) < val) {
    int assumed = old;
    old = atomicCAS(ia, assumed, __float_as_int(val));
    if (old == assumed) break;
  }
}

__global__ void seg_max_kernel(const float* __restrict__ logits, const int* __restrict__ batch,
                               float* m) {
  int n = blockIdx.x * blockDim.x + threadIdx.x;
  if (n < NN) atomicMaxF(&m[batch[n]], logits[n]);
}

__global__ void exp_sum_kernel(const float* __restrict__ logits, const float* __restrict__ m,
                               const int* __restrict__ batch, float* __restrict__ e, float* s) {
  int n = blockIdx.x * blockDim.x + threadIdx.x;
  if (n >= NN) return;
  int b = batch[n];
  float ev = __expf(logits[n] - m[b]);
  e[n] = ev;
  atomicAdd(&s[b], ev);
}

__global__ void attn_norm_kernel(const float* __restrict__ e, const float* __restrict__ s,
                                 const int* __restrict__ batch, float* __restrict__ attn) {
  int n = blockIdx.x * blockDim.x + threadIdx.x;
  if (n >= NN) return;
  attn[n] = e[n] / (s[batch[n]] + 1e-9f);
}

__global__ void upd_scatter_kernel(const float* __restrict__ attn, const float* __restrict__ v,
                                   const int* __restrict__ batch, float* upd) {
  int t = blockIdx.x * blockDim.x + threadIdx.x;   // NN*16 float4 chunks
  if (t >= NN * 16) return;
  int n = t >> 4;
  int c = (t & 15) << 2;
  int b = batch[n];
  float a = attn[n];
  float4 vv = *(const float4*)(v + (size_t)n * DD + c);
  float* base = upd + (size_t)b * DD + c;
  atomicAdd(base + 0, a * vv.x); atomicAdd(base + 1, a * vv.y);
  atomicAdd(base + 2, a * vv.z); atomicAdd(base + 3, a * vv.w);
}

// out[g,j] = sum_k U[g,k]*Wt[j,k] + bias[j]   (Wt is [192][64], i.e. x @ W.T)
__global__ void gru_linear_kernel(const float* __restrict__ U, const float* __restrict__ Wt,
                                  const float* __restrict__ bias, float* __restrict__ out) {
  int t = blockIdx.x * blockDim.x + threadIdx.x;   // GG*TD
  if (t >= GG * TD) return;
  int g = t / TD, j = t - g * TD;
  const float* u = U + (size_t)g * DD;
  const float* w = Wt + (size_t)j * DD;
  float acc = bias[j];
#pragma unroll 8
  for (int k = 0; k < DD; ++k) acc += u[k] * w[k];
  out[t] = acc;
}

__global__ void gru_gate_kernel(const float* __restrict__ gi, const float* __restrict__ gh,
                                float* slots) {
  int t = blockIdx.x * blockDim.x + threadIdx.x;   // GG*DD
  if (t >= GG * DD) return;
  int g = t >> 6, d = t & 63;
  const float* gir = gi + (size_t)g * TD;
  const float* ghr = gh + (size_t)g * TD;
  float r  = 1.f / (1.f + __expf(-(gir[d]       + ghr[d])));
  float zz = 1.f / (1.f + __expf(-(gir[DD + d]  + ghr[DD + d])));
  float nn = tanhf(gir[2*DD + d] + r * ghr[2*DD + d]);
  slots[t] = (1.f - zz) * nn + zz * slots[t];
}

__global__ void noisy_kernel(const float* __restrict__ attn, const float* __restrict__ h,
                             const float* __restrict__ noise, float* __restrict__ out) {
  int t = blockIdx.x * blockDim.x + threadIdx.x;   // NN*DD
  if (t >= NN * DD) return;
  out[t] = attn[t >> 6] * h[t] + noise[t];
}

// ----------------------------------------------------------------------------
extern "C" void kernel_launch(void* const* d_in, const int* in_sizes, int n_in,
                              void* d_out, int out_size, void* d_ws, size_t ws_size,
                              hipStream_t stream) {
  (void)in_sizes; (void)n_in; (void)out_size; (void)ws_size;
  const float* x        = (const float*)d_in[0];
  const int*   ei       = (const int*)d_in[1];
  const int*   batch    = (const int*)d_in[2];
  const float* noise    = (const float*)d_in[3];
  const float* W1_0     = (const float*)d_in[4];
  const float* b1_0     = (const float*)d_in[5];
  const float* W2_0     = (const float*)d_in[6];
  const float* b2_0     = (const float*)d_in[7];
  const float* W1_r     = (const float*)d_in[8];
  const float* b1_r     = (const float*)d_in[9];
  const float* W2_r     = (const float*)d_in[10];
  const float* b2_r     = (const float*)d_in[11];
  const float* bn_g     = (const float*)d_in[12];
  const float* bn_b     = (const float*)d_in[13];
  const float* Wq       = (const float*)d_in[14];
  const float* Wk       = (const float*)d_in[15];
  const float* Wv       = (const float*)d_in[16];
  const float* gWih     = (const float*)d_in[17];
  const float* gWhh     = (const float*)d_in[18];
  const float* gbih     = (const float*)d_in[19];
  const float* gbhh     = (const float*)d_in[20];
  const float* nmuW     = (const float*)d_in[21];
  const float* nmub     = (const float*)d_in[22];
  const float* nlvW     = (const float*)d_in[23];
  const float* nlvb     = (const float*)d_in[24];
  const float* gmuW     = (const float*)d_in[25];
  const float* gmub     = (const float*)d_in[26];
  const float* glvW     = (const float*)d_in[27];
  const float* glvb     = (const float*)d_in[28];
  const float* pg       = (const float*)d_in[29];
  const float* pb       = (const float*)d_in[30];
  const int* srcI = ei;
  const int* dstI = ei + EE;
  float* out = (float*)d_out;

  // ---- workspace layout (floats) ----
  float* W = (float*)d_ws;
  const size_t N64 = (size_t)NN * DD;
  float* agg   = W;                       // NN*128 (layer0 agg; later reused as noisy)
  float* h     = agg  + (size_t)NN * FF;  // NN*64
  float* t1    = h    + N64;              // NN*64
  float* t2    = t1   + N64;              // NN*64
  float* kbuf  = t2   + N64;              // NN*64
  float* vbuf  = kbuf + N64;              // NN*64
  float* logit = vbuf + N64;              // NN
  float* ebuf  = logit + NN;              // NN
  float* attn  = ebuf + NN;               // NN
  float* cnt   = attn + NN;               // GG
  float* mseg  = cnt + GG;                // GG
  float* sseg  = mseg + GG;               // GG
  float* slots = sseg + GG;               // GG*64
  float* qbuf  = slots + (size_t)GG * DD; // GG*64
  float* upd   = qbuf  + (size_t)GG * DD; // GG*64
  float* gi    = upd   + (size_t)GG * DD; // GG*192
  float* gh    = gi    + (size_t)GG * TD; // GG*192
  float* stats = gh    + (size_t)GG * TD; // 128

  const int BS = 256;
  auto nb = [](long long n, int bs) { return (unsigned)((n + bs - 1) / bs); };
  dim3 gemmBlkN((NN / 16 + 7) / 8);       // 391 blocks of 8 waves
  dim3 gemmBlkG((GG / 16 + 7) / 8);       // 4 blocks

  // -------------------- GIN layer 0 (K = 128) --------------------
  copy4_kernel<<<nb((long long)NN * FF / 4, BS), BS, 0, stream>>>(x, agg, NN * FF / 4);
  edge_scatter_kernel<<<nb((long long)EE * (FF / 4), BS), BS, 0, stream>>>(x, srcI, dstI, agg, FF);
  gemm64_kernel<<<gemmBlkN, 256, 0, stream>>>(agg, W1_0, b1_0, t1, NN, FF, 1);
  gemm64_kernel<<<gemmBlkN, 256, 0, stream>>>(t1, W2_0, b2_0, t2, NN, DD, 0);
  fill_kernel<<<1, 128, 0, stream>>>(stats, 0.f, 128);
  bn_reduce_kernel<<<256, 256, 0, stream>>>(t2, stats, NN);
  bn_apply_kernel<<<nb(N64, BS), BS, 0, stream>>>(t2, stats, bn_g + 0 * DD, bn_b + 0 * DD, h, NN);

  // -------------------- GIN layers 1..2 (K = 64) --------------------
  for (int i = 0; i < 2; ++i) {
    copy4_kernel<<<nb(N64 / 4, BS), BS, 0, stream>>>(h, agg, (int)(N64 / 4));
    edge_scatter_kernel<<<nb((long long)EE * (DD / 4), BS), BS, 0, stream>>>(h, srcI, dstI, agg, DD);
    gemm64_kernel<<<gemmBlkN, 256, 0, stream>>>(agg, W1_r + (size_t)i * DD * DD, b1_r + i * DD, t1, NN, DD, 1);
    gemm64_kernel<<<gemmBlkN, 256, 0, stream>>>(t1, W2_r + (size_t)i * DD * DD, b2_r + i * DD, t2, NN, DD, 0);
    fill_kernel<<<1, 128, 0, stream>>>(stats, 0.f, 128);
    bn_reduce_kernel<<<256, 256, 0, stream>>>(t2, stats, NN);
    bn_apply_kernel<<<nb(N64, BS), BS, 0, stream>>>(t2, stats, bn_g + (i + 1) * DD, bn_b + (i + 1) * DD, h, NN);
  }

  // -------------------- summary maker --------------------
  fill_kernel<<<nb(GG, BS), BS, 0, stream>>>(cnt, 0.f, GG);
  fill_kernel<<<nb(GG * DD, BS), BS, 0, stream>>>(slots, 0.f, GG * DD);
  count_nodes_kernel<<<nb(NN, BS), BS, 0, stream>>>(batch, cnt);
  slot_sum_kernel<<<nb((long long)NN * 16, BS), BS, 0, stream>>>(h, batch, slots);
  slot_div_kernel<<<nb(GG * DD, BS), BS, 0, stream>>>(slots, cnt);
  gemm64_kernel<<<gemmBlkN, 256, 0, stream>>>(h, Wk, nullptr, kbuf, NN, DD, 0);
  gemm64_kernel<<<gemmBlkN, 256, 0, stream>>>(h, Wv, nullptr, vbuf, NN, DD, 0);

  for (int it = 0; it < 2; ++it) {
    gemm64_kernel<<<gemmBlkG, 256, 0, stream>>>(slots, Wq, nullptr, qbuf, GG, DD, 0);
    attn_logits_kernel<<<nb(NN, BS), BS, 0, stream>>>(kbuf, qbuf, batch, logit);
    fill_kernel<<<nb(GG, BS), BS, 0, stream>>>(mseg, -3.0e38f, GG);
    fill_kernel<<<nb(GG, BS), BS, 0, stream>>>(sseg, 0.f, GG);
    seg_max_kernel<<<nb(NN, BS), BS, 0, stream>>>(logit, batch, mseg);
    exp_sum_kernel<<<nb(NN, BS), BS, 0, stream>>>(logit, mseg, batch, ebuf, sseg);
    attn_norm_kernel<<<nb(NN, BS), BS, 0, stream>>>(ebuf, sseg, batch, attn);
    fill_kernel<<<nb(GG * DD, BS), BS, 0, stream>>>(upd, 0.f, GG * DD);
    upd_scatter_kernel<<<nb((long long)NN * 16, BS), BS, 0, stream>>>(attn, vbuf, batch, upd);
    gru_linear_kernel<<<nb(GG * TD, BS), BS, 0, stream>>>(upd, gWih, gbih, gi);
    gru_linear_kernel<<<nb(GG * TD, BS), BS, 0, stream>>>(slots, gWhh, gbhh, gh);
    gru_gate_kernel<<<nb(GG * DD, BS), BS, 0, stream>>>(gi, gh, slots);
  }

  float* noisy = agg;   // reuse big buffer
  noisy_kernel<<<nb(N64, BS), BS, 0, stream>>>(attn, h, noise, noisy);

  // -------------------- projection heads + BN --------------------
  // node_mu -> out[0 : N*64)
  gemm64_kernel<<<gemmBlkN, 256, 0, stream>>>(noisy, nmuW, nmub, t1, NN, DD, 0);
  fill_kernel<<<1, 128, 0, stream>>>(stats, 0.f, 128);
  bn_reduce_kernel<<<256, 256, 0, stream>>>(t1, stats, NN);
  bn_apply_kernel<<<nb(N64, BS), BS, 0, stream>>>(t1, stats, pg + 0 * DD, pb + 0 * DD, out, NN);
  // node_lv -> out[N*64 : 2*N*64)
  gemm64_kernel<<<gemmBlkN, 256, 0, stream>>>(noisy, nlvW, nlvb, t1, NN, DD, 0);
  fill_kernel<<<1, 128, 0, stream>>>(stats, 0.f, 128);
  bn_reduce_kernel<<<256, 256, 0, stream>>>(t1, stats, NN);
  bn_apply_kernel<<<nb(N64, BS), BS, 0, stream>>>(t1, stats, pg + 1 * DD, pb + 1 * DD, out + N64, NN);
  // graph_mu -> out[2*N*64 : 2*N*64 + G*64)
  gemm64_kernel<<<gemmBlkG, 256, 0, stream>>>(slots, gmuW, gmub, t1, GG, DD, 0);
  fill_kernel<<<1, 128, 0, stream>>>(stats, 0.f, 128);
  bn_reduce_kernel<<<128, 256, 0, stream>>>(t1, stats, GG);
  bn_apply_kernel<<<nb(GG * DD, BS), BS, 0, stream>>>(t1, stats, pg + 2 * DD, pb + 2 * DD, out + 2 * N64, GG);
  // graph_lv -> out[... + G*64 : ... + 2*G*64)
  gemm64_kernel<<<gemmBlkG, 256, 0, stream>>>(slots, glvW, glvb, t1, GG, DD, 0);
  fill_kernel<<<1, 128, 0, stream>>>(stats, 0.f, 128);
  bn_reduce_kernel<<<128, 256, 0, stream>>>(t1, stats, GG);
  bn_apply_kernel<<<nb(GG * DD, BS), BS, 0, stream>>>(t1, stats, pg + 3 * DD, pb + 3 * DD,
                                                      out + 2 * N64 + (size_t)GG * DD, GG);
}
